// PointnetFPModule_63178968924297
// MI455X (gfx1250) — matrix operand order repaired and checked by reading, not compiled, not run on hardware
//
#include <hip/hip_runtime.h>
#include <stdint.h>

// Problem constants (from reference setup_inputs)
#define NPTS 131072
#define CCH  512
#define NSEG 16

typedef __attribute__((ext_vector_type(16))) __bf16 v16bf;
typedef __attribute__((ext_vector_type(8)))  float  v8f;
typedef __attribute__((ext_vector_type(4)))  unsigned int v4ui;
typedef __attribute__((ext_vector_type(8)))  int v8i;
typedef __attribute__((ext_vector_type(4)))  int v4i;

union BFrag { uint4 u[2]; v16bf v; };

__device__ __forceinline__ unsigned short f2bf(float f) {
  unsigned int u = __float_as_uint(f);
  unsigned int r = 0x7FFFu + ((u >> 16) & 1u);
  return (unsigned short)((u + r) >> 16);
}
__device__ __forceinline__ unsigned int pk2(float a, float b) {
  return (unsigned int)f2bf(a) | ((unsigned int)f2bf(b) << 16);
}

// ---- Tensor Data Mover: DMA a 128-row x 32-col fp32 tile (row stride 512) ----
// D# per CDNA5 ISA:
//  group0: [1:0]=count=1 | [63:32]=lds_addr | [120:64]=global_addr | [127:126]=type=2
//  group1: [17:16]=data_size(2 -> 4B) | [79:48]=tensor_dim0=512 |
//          [111:80]=tensor_dim1=131072 | [127:112]=tile_dim0=32 |
//          [143:128]=tile_dim1=128 | [207:160]=tensor_dim0_stride=512
__device__ __forceinline__ void tdm_load_tile(unsigned lds_addr, const float* gsrc) {
  unsigned long long ga = (unsigned long long)(uintptr_t)gsrc;
  v4ui g0;
  g0.x = 1u;                                                // count = 1 (valid)
  g0.y = lds_addr;                                          // LDS byte address
  g0.z = (unsigned)(ga & 0xFFFFFFFFu);                      // global_addr[31:0]
  g0.w = (unsigned)((ga >> 32) & 0x1FFFFFFu) | (2u << 30);  // global_addr[56:32], type=2
  v8i g1;
  g1[0] = (int)(2u << 16);                                  // data_size = 4 bytes
  g1[1] = (int)(512u << 16);                                // tensor_dim0 = 512 (lo16)
  g1[2] = 0;                                                // dim0 hi, dim1 lo16 (131072 -> 0)
  g1[3] = (int)((32u << 16) | 2u);                          // dim1 hi = 2, tile_dim0 = 32
  g1[4] = 128;                                              // tile_dim1 = 128, tile_dim2 = 0
  g1[5] = 512;                                              // tensor_dim0_stride = 512
  g1[6] = 0;
  g1[7] = 0;
  v4i zz = {0, 0, 0, 0};
#if __clang_major__ >= 23
  v8i zz8 = {0, 0, 0, 0, 0, 0, 0, 0};
  __builtin_amdgcn_tensor_load_to_lds(g0, g1, zz, zz, zz8, 0);
#else
  __builtin_amdgcn_tensor_load_to_lds(g0, g1, zz, zz, 0);
#endif
}

// ---------------- workspace layout (bytes) ----------------
#define OFF_SUMS    0
#define OFF_BNSUM   32768
#define OFF_BNSQ    34816
#define OFF_SCSH    36864
#define OFF_PROJB   40960
#define OFF_PROJBF  73728
#define OFF_WBP     90112
#define OFF_WB2P    614400
#define OFF_WB1BP   1138688

// ---------------- zero accumulators ----------------
__global__ void k_zero(float* ws) {
  int i = blockIdx.x * 256 + threadIdx.x;   // 9216 floats: sums + bnsum + bnsumsq
  if (i < 9216) ws[i] = 0.0f;
}

// ---------------- pack weights into WMMA B-fragment order ----------------
// Packed index: idx = ((kt*32 + nt)*32 + lane)*16 + i
//   n = nt*16 + (lane&15);  k = kt*32 + ((i>>3)<<4) + ((lane>>4)<<3) + (i&7)
__global__ void k_pack(const float* __restrict__ w1, const float* __restrict__ w2,
                       unsigned short* __restrict__ wBp,
                       unsigned short* __restrict__ wB2p,
                       unsigned short* __restrict__ wB1bp) {
  int id  = blockIdx.x * 256 + threadIdx.x;   // 3 * 262144
  int mat = id >> 18;
  int idx = id & 262143;
  int kt  = idx >> 14;
  int rem = idx & 16383;
  int nt  = rem >> 9;
  int l   = (rem >> 4) & 31;
  int i   = rem & 15;
  int n   = nt * 16 + (l & 15);
  int k   = kt * 32 + ((i >> 3) << 4) + ((l >> 4) << 3) + (i & 7);
  float v;
  if (mat == 0)      v = w1[n * 1024 + k];
  else if (mat == 1) v = w2[n * 512 + k];
  else               v = w1[n * 1024 + 512 + k];
  unsigned short b = f2bf(v);
  if (mat == 0)      wBp[idx]   = b;
  else if (mat == 1) wB2p[idx]  = b;
  else               wB1bp[idx] = b;
}

// ---------------- ragged segment sums ----------------
__global__ void k_segsum(const float* __restrict__ x, const int* __restrict__ o,
                         float* __restrict__ sums) {
  __shared__ int so[NSEG];
  int tid = threadIdx.x;
  if (tid < NSEG) so[tid] = o[tid];
  __syncthreads();
  int r0 = blockIdx.x * 128;
  int c  = tid * 2;
  int seg = 0;
  while (seg < NSEG - 1 && r0 >= so[seg]) seg++;
  int cur = seg;
  float a0 = 0.f, a1 = 0.f;
  for (int j = 0; j < 128; ++j) {
    int row = r0 + j;
    while (seg < NSEG - 1 && row >= so[seg]) seg++;
    if (seg != cur) {
      atomicAdd(&sums[cur * CCH + c], a0);
      atomicAdd(&sums[cur * CCH + c + 1], a1);
      a0 = a1 = 0.f; cur = seg;
    }
    float2 v = *(const float2*)(x + (size_t)row * CCH + c);
    a0 += v.x; a1 += v.y;
  }
  atomicAdd(&sums[cur * CCH + c], a0);
  atomicAdd(&sums[cur * CCH + c + 1], a1);
}

// ---------------- means -> proj = ReLU(means @ w2^T + b2), bf16 out ----------------
__global__ void k_proj(const float* __restrict__ sums, const int* __restrict__ o,
                       const unsigned short* __restrict__ wB2p,
                       const float* __restrict__ b2,
                       unsigned short* __restrict__ projbf) {
  __shared__ unsigned short Asm[NSEG * CCH];  // 16 KB
  int tid = threadIdx.x;
  for (int j = 0; j < 32; ++j) {
    int idx = tid + j * 256;                  // 0..8191
    int s   = idx >> 9;
    int prev = (s == 0) ? 0 : o[s - 1];
    float cnt = (float)(o[s] - prev);
    Asm[idx] = f2bf(sums[idx] / cnt);
  }
  __syncthreads();
  int wave = tid >> 5, lane = tid & 31;
  v8f acc[4] = {};
  for (int kt = 0; kt < 16; ++kt) {
    int arow = lane & 15;
    int aoff = arow * CCH + kt * 32 + ((lane >> 4) << 3);
    BFrag a;
    a.u[0] = *(const uint4*)&Asm[aoff];
    a.u[1] = *(const uint4*)&Asm[aoff + 16];
    for (int t = 0; t < 4; ++t) {
      int ntg = wave * 4 + t;
      const uint4* bp = (const uint4*)(wB2p + (size_t)((kt * 32 + ntg) * 32 + lane) * 16);
      BFrag b; b.u[0] = bp[0]; b.u[1] = bp[1];
      acc[t] = __builtin_amdgcn_wmma_f32_16x16x32_bf16(false, a.v, false, b.v,
                                                       (short)0, acc[t], false, false);
    }
  }
  for (int t = 0; t < 4; ++t) {
    int col = (wave * 4 + t) * 16 + (lane & 15);
    float bb = b2[col];
    for (int r = 0; r < 8; ++r) {
      int s = r + ((lane >> 4) << 3);
      float val = acc[t][r] + bb;
      val = fmaxf(val, 0.0f);
      projbf[s * CCH + col] = f2bf(val);
    }
  }
}

// ---------------- projb = proj @ w1[:,512:]^T + b1 (f32 out) ----------------
__global__ void k_projb(const unsigned short* __restrict__ projbf,
                        const unsigned short* __restrict__ wB1bp,
                        const float* __restrict__ b1,
                        float* __restrict__ projb) {
  __shared__ unsigned short Asm[NSEG * CCH];  // 16 KB
  int tid = threadIdx.x;
  for (int j = 0; j < 16; ++j) {
    int idx = tid + j * 256;                  // 4096 uints
    ((unsigned int*)Asm)[idx] = ((const unsigned int*)projbf)[idx];
  }
  __syncthreads();
  int wave = tid >> 5, lane = tid & 31;
  v8f acc[4] = {};
  for (int kt = 0; kt < 16; ++kt) {
    int arow = lane & 15;
    int aoff = arow * CCH + kt * 32 + ((lane >> 4) << 3);
    BFrag a;
    a.u[0] = *(const uint4*)&Asm[aoff];
    a.u[1] = *(const uint4*)&Asm[aoff + 16];
    for (int t = 0; t < 4; ++t) {
      int ntg = wave * 4 + t;
      const uint4* bp = (const uint4*)(wB1bp + (size_t)((kt * 32 + ntg) * 32 + lane) * 16);
      BFrag b; b.u[0] = bp[0]; b.u[1] = bp[1];
      acc[t] = __builtin_amdgcn_wmma_f32_16x16x32_bf16(false, a.v, false, b.v,
                                                       (short)0, acc[t], false, false);
    }
  }
  for (int t = 0; t < 4; ++t) {
    int col = (wave * 4 + t) * 16 + (lane & 15);
    float bb = b1[col];
    for (int r = 0; r < 8; ++r) {
      int s = r + ((lane >> 4) << 3);
      projb[s * CCH + col] = acc[t][r] + bb;
    }
  }
}

// ---------------- main GEMM: h = x @ w1[:, :512]^T + projb[seg]  (+ BN partials) ----
// Grid: (NPTS/128, 512/128). Block 256 thr = 8 waves; wave = (rw 0..3) x (cw 0..1),
// each wave: 2 row-tiles x 4 col-tiles of 16x16, K-loop 512/32.
// x tiles are DMA'd into LDS by the Tensor Data Mover, double-buffered:
// wave 0 issues tile kt+1 while all waves compute tile kt.
__global__ void k_gemm(const float* __restrict__ x, const int* __restrict__ o,
                       const unsigned short* __restrict__ wBp,
                       const float* __restrict__ projb,
                       float* __restrict__ out,
                       float* __restrict__ bnsum, float* __restrict__ bnsumsq) {
  __shared__ float Xs[2][128 * 32];   // 32 KB fp32 x-tile, double buffered
  __shared__ float sm_bn[256];
  __shared__ int segrow[128];
  int tid = threadIdx.x;
  int r0 = blockIdx.x * 128;
  int c0 = blockIdx.y * 128;
  sm_bn[tid] = 0.0f;
  if (tid < 128) {
    int row = r0 + tid, s = 0;
    while (s < NSEG - 1 && row >= o[s]) s++;
    segrow[tid] = s;
  }
  int wave = tid >> 5, lane = tid & 31;
  int rw = wave & 3, cw = wave >> 2;
  v8f acc[2][4] = {};
  int nt0 = (c0 >> 4) + cw * 4;
  unsigned ldsXs0 = (unsigned)(uintptr_t)&Xs[0][0];
  unsigned ldsXs1 = (unsigned)(uintptr_t)&Xs[1][0];
  const float* xtile = x + (size_t)r0 * CCH;

  if (wave == 0) tdm_load_tile(ldsXs0, xtile);          // prologue: tile 0

  for (int kt = 0; kt < 16; ++kt) {
    if (wave == 0) __builtin_amdgcn_s_wait_tensorcnt(0);
    __syncthreads();                                    // tile kt visible to all waves
    if (wave == 0 && kt < 15)                           // DMA tile kt+1 into other buffer
      tdm_load_tile((kt & 1) ? ldsXs0 : ldsXs1, xtile + (kt + 1) * 32);
    if (kt < 15)
      __builtin_prefetch(wBp + (size_t)(((kt + 1) * 32 + nt0) * 32) * 16, 0, 1);

    const float* xs = &Xs[kt & 1][0];
    // Build A fragments (two 16-row tiles): fp32 from LDS -> packed bf16
    BFrag a[2];
    for (int rt = 0; rt < 2; ++rt) {
      int row = rw * 32 + rt * 16 + (lane & 15);
      int kb  = (lane >> 4) << 3;
      const float* base = xs + row * 32;
      float4 q0 = *(const float4*)(base + kb);
      float4 q1 = *(const float4*)(base + kb + 4);
      float4 q2 = *(const float4*)(base + 16 + kb);
      float4 q3 = *(const float4*)(base + 16 + kb + 4);
      a[rt].u[0] = make_uint4(pk2(q0.x, q0.y), pk2(q0.z, q0.w),
                              pk2(q1.x, q1.y), pk2(q1.z, q1.w));
      a[rt].u[1] = make_uint4(pk2(q2.x, q2.y), pk2(q2.z, q2.w),
                              pk2(q3.x, q3.y), pk2(q3.z, q3.w));
    }
    for (int t2 = 0; t2 < 4; ++t2) {
      int ntg = nt0 + t2;
      const uint4* bp = (const uint4*)(wBp + (size_t)((kt * 32 + ntg) * 32 + lane) * 16);
      BFrag b; b.u[0] = bp[0]; b.u[1] = bp[1];
      acc[0][t2] = __builtin_amdgcn_wmma_f32_16x16x32_bf16(false, a[0].v, false, b.v,
                                                           (short)0, acc[0][t2], false, false);
      acc[1][t2] = __builtin_amdgcn_wmma_f32_16x16x32_bf16(false, a[1].v, false, b.v,
                                                           (short)0, acc[1][t2], false, false);
    }
  }
  __syncthreads();
  // epilogue: + projb[seg], store pre-BN h, accumulate per-channel sum/sumsq
  for (int rt = 0; rt < 2; ++rt) {
    for (int t2 = 0; t2 < 4; ++t2) {
      int lc  = cw * 64 + t2 * 16 + (lane & 15);
      int col = c0 + lc;
      float s1 = 0.f, s2 = 0.f;
      for (int r = 0; r < 8; ++r) {
        int lrow = rw * 32 + rt * 16 + r + ((lane >> 4) << 3);
        float val = acc[rt][t2][r] + projb[segrow[lrow] * CCH + col];
        out[(size_t)(r0 + lrow) * CCH + col] = val;
        s1 += val; s2 += val * val;
      }
      atomicAdd(&sm_bn[lc], s1);
      atomicAdd(&sm_bn[128 + lc], s2);
    }
  }
  __syncthreads();
  if (tid < 128) {
    atomicAdd(&bnsum[c0 + tid],   sm_bn[tid]);
    atomicAdd(&bnsumsq[c0 + tid], sm_bn[128 + tid]);
  }
}

// ---------------- finalize BN scale/shift ----------------
__global__ void k_stats(const float* __restrict__ bnsum, const float* __restrict__ bnsumsq,
                        const float* __restrict__ gamma, const float* __restrict__ beta,
                        float* __restrict__ scsh) {
  int c = blockIdx.x * 256 + threadIdx.x;
  if (c < CCH) {
    float inv = 1.0f / (float)NPTS;
    float mu  = bnsum[c] * inv;
    float var = bnsumsq[c] * inv - mu * mu;
    float rs  = rsqrtf(var + 1e-5f);
    float sc  = gamma[c] * rs;
    scsh[c]       = sc;
    scsh[CCH + c] = beta[c] - mu * sc;
  }
}

// ---------------- BN apply + ReLU, in place on d_out ----------------
__global__ void k_bnrelu(float* __restrict__ out, const float* __restrict__ scsh) {
  size_t i4 = (size_t)blockIdx.x * 256 + threadIdx.x;   // NPTS*CCH/4 elements
  float4 v = ((const float4*)out)[i4];
  int cb = (int)((i4 * 4) & (CCH - 1));
  float4 sc = *(const float4*)(scsh + cb);
  float4 sh = *(const float4*)(scsh + CCH + cb);
  v.x = fmaxf(v.x * sc.x + sh.x, 0.f);
  v.y = fmaxf(v.y * sc.y + sh.y, 0.f);
  v.z = fmaxf(v.z * sc.z + sh.z, 0.f);
  v.w = fmaxf(v.w * sc.w + sh.w, 0.f);
  ((float4*)out)[i4] = v;
}

extern "C" void kernel_launch(void* const* d_in, const int* in_sizes, int n_in,
                              void* d_out, int out_size, void* d_ws, size_t ws_size,
                              hipStream_t stream) {
  const float* x     = (const float*)d_in[0];
  const int*   o     = (const int*)d_in[1];
  const float* w2    = (const float*)d_in[2];
  const float* b2    = (const float*)d_in[3];
  const float* w1    = (const float*)d_in[4];
  const float* b1    = (const float*)d_in[5];
  const float* gamma = (const float*)d_in[6];
  const float* beta  = (const float*)d_in[7];
  float* out = (float*)d_out;

  char* ws = (char*)d_ws;
  float*          sums    = (float*)(ws + OFF_SUMS);
  float*          bnsum   = (float*)(ws + OFF_BNSUM);
  float*          bnsumsq = (float*)(ws + OFF_BNSQ);
  float*          scsh    = (float*)(ws + OFF_SCSH);
  float*          projb   = (float*)(ws + OFF_PROJB);
  unsigned short* projbf  = (unsigned short*)(ws + OFF_PROJBF);
  unsigned short* wBp     = (unsigned short*)(ws + OFF_WBP);
  unsigned short* wB2p    = (unsigned short*)(ws + OFF_WB2P);
  unsigned short* wB1bp   = (unsigned short*)(ws + OFF_WB1BP);

  k_zero<<<36, 256, 0, stream>>>((float*)ws);
  k_pack<<<3072, 256, 0, stream>>>(w1, w2, wBp, wB2p, wB1bp);
  k_segsum<<<NPTS / 128, 256, 0, stream>>>(x, o, sums);
  k_proj<<<1, 256, 0, stream>>>(sums, o, wB2p, b2, projbf);
  k_projb<<<1, 256, 0, stream>>>(projbf, wB1bp, b1, projb);
  k_gemm<<<dim3(NPTS / 128, CCH / 128), 256, 0, stream>>>(x, o, wBp, projb, out,
                                                          bnsum, bnsumsq);
  k_stats<<<2, 256, 0, stream>>>(bnsum, bnsumsq, gamma, beta, scsh);
  k_bnrelu<<<(NPTS * CCH) / (256 * 4), 256, 0, stream>>>(out, scsh);
}
